// VisualBranch_vsgnet_87162066305839
// MI455X (gfx1250) — matrix operand order repaired and checked by reading, not compiled, not run on hardware
//
#include <hip/hip_runtime.h>
#include <hip/hip_bf16.h>

typedef __attribute__((ext_vector_type(4)))  __bf16 v4bf;
typedef __attribute__((ext_vector_type(8)))  __bf16 v8bf;
typedef __attribute__((ext_vector_type(16))) __bf16 v16bf;
typedef __attribute__((ext_vector_type(8)))  float  v8f;
typedef __attribute__((ext_vector_type(4)))  int    v4i;

// ---------------- constants ----------------
#define BATCH      32
#define NREL       512
#define MROWS      16384          // 32*512 relations
#define DIM        1024
#define OBJ_ROWS   3200           // 32*100
#define OUT_OFS    3276800        // OBJ_ROWS*DIM floats
#define LDH        40             // LDS row pitch in halves (80B, 16B aligned, conflict-free)

// ---- CDNA5 async global->LDS path (direct, ASYNCcnt-tracked), with fallback ----
#if defined(__has_builtin)
#if __has_builtin(__builtin_amdgcn_global_load_async_to_lds_b128)
#define HAVE_ASYNC_LDS 1
#endif
#endif

#ifdef HAVE_ASYNC_LDS
__device__ __forceinline__ void async_cp16(const __bf16* g, __bf16* l) {
  __builtin_amdgcn_global_load_async_to_lds_b128(
      (__attribute__((address_space(1))) v4i*)(g),   // global src (as1)
      (__attribute__((address_space(3))) v4i*)(l),   // LDS dst (as3)
      0, 0);
}
#if __has_builtin(__builtin_amdgcn_s_wait_asynccnt)
#define WAIT_ASYNC0() __builtin_amdgcn_s_wait_asynccnt(0)
#else
#define WAIT_ASYNC0() asm volatile("s_wait_asynccnt 0x0" ::: "memory")
#endif
#endif

__device__ __forceinline__ v16bf cat8(v8bf lo, v8bf hi) {
  return __builtin_shufflevector(lo, hi, 0,1,2,3,4,5,6,7,8,9,10,11,12,13,14,15);
}

// ---------------- 1) copy object_branch_output to d_out ----------------
__global__ __launch_bounds__(256) void copy_obj_kernel(const float4* __restrict__ src,
                                                       float4* __restrict__ dst) {
  int i = blockIdx.x * 256 + threadIdx.x;
  dst[i] = src[i];
}

// ---------------- 2) transpose fp32 [K][N] -> bf16 [N][K] ----------------
__global__ __launch_bounds__(256) void transpose_to_bf16(const float* __restrict__ src,
                                                         __bf16* __restrict__ dst,
                                                         int K, int N) {
  __shared__ float tile[64][65];
  int kb = blockIdx.x * 64, nb = blockIdx.y * 64;
  int t = threadIdx.x;
#pragma unroll
  for (int i = 0; i < 16; ++i) {
    int idx = i * 256 + t;
    int r = idx >> 6, c = idx & 63;
    tile[r][c] = src[(size_t)(kb + r) * N + (nb + c)];
  }
  __syncthreads();
#pragma unroll
  for (int i = 0; i < 16; ++i) {
    int idx = i * 256 + t;
    int n = idx >> 6, k = idx & 63;
    dst[(size_t)(nb + n) * K + (kb + k)] = (__bf16)tile[k][n];
  }
}

// ---------------- 3) ctxc[b][j] = b1[j] + sum_k ctx[b][k]*W1[1024+k][j] ----------------
__global__ __launch_bounds__(256) void ctx_contrib_kernel(const float* __restrict__ ctx,
                                                          const float* __restrict__ W1,
                                                          const float* __restrict__ b1,
                                                          float* __restrict__ ctxc) {
  int j = blockIdx.x * 256 + threadIdx.x;
  int b = blockIdx.y;
  const float* w = W1 + (size_t)DIM * DIM + j;   // rows 1024..2047, column j
  const float* c = ctx + (size_t)b * DIM;
  float sum = b1[j];
#pragma unroll 4
  for (int k = 0; k < DIM; ++k) sum = fmaf(c[k], w[(size_t)k * DIM], sum);
  ctxc[(size_t)b * DIM + j] = sum;
}

// ---------------- 4) gather pair endpoints, mean, convert to bf16 ----------------
__global__ __launch_bounds__(256) void gather_pair_mean(const float* __restrict__ obj,
                                                        const int* __restrict__ num_obj,
                                                        const int* __restrict__ pairs,
                                                        __bf16* __restrict__ V) {
  int row = blockIdx.x;            // 0..16383
  int b = row >> 9;
  int r = row & (NREL - 1);
  int off = 0;
  for (int i = 0; i < b; ++i) off += num_obj[i];   // exclusive cumsum (uniform per block)
  int p0 = pairs[(size_t)((b << 9) + r) * 2 + 0] + off;
  int p1 = pairs[(size_t)((b << 9) + r) * 2 + 1] + off;
  int col = threadIdx.x * 4;
  float4 a = *(const float4*)(obj + (size_t)p0 * DIM + col);
  float4 c = *(const float4*)(obj + (size_t)p1 * DIM + col);
  v4bf o;
  o[0] = (__bf16)((a.x + c.x) * 0.5f);
  o[1] = (__bf16)((a.y + c.y) * 0.5f);
  o[2] = (__bf16)((a.z + c.z) * 0.5f);
  o[3] = (__bf16)((a.w + c.w) * 0.5f);
  *(v4bf*)(V + (size_t)row * DIM + col) = o;
}

// ---------------- 5) bf16 WMMA GEMM: C = relu(A[M,1024] @ Bt^T + add) ----------------
// A row-major [M][1024] bf16. Bt is B transposed: [N][K] bf16 (K contiguous).
// MODE 0: add = ctxc[row>>9][col], output bf16 -> outH.   MODE 1: add = bias[col], output f32 -> outF.
template <int MODE>
__global__ __launch_bounds__(256)
void gemm_bf16_wmma(const __bf16* __restrict__ A, const __bf16* __restrict__ Bt,
                    const float* __restrict__ add,
                    __bf16* __restrict__ outH, float* __restrict__ outF) {
  __shared__ __bf16 sA[2][128 * LDH];
  __shared__ __bf16 sB[2][128 * LDH];

  const int t    = threadIdx.x;
  const int lane = t & 31;
  const int wave = t >> 5;
  const int wm   = wave & 3;   // 4 waves along M (32 rows each)
  const int wn   = wave >> 2;  // 2 waves along N (64 cols each)
  const int mBase = blockIdx.x * 128;
  const int nBase = blockIdx.y * 128;

  // global staging: 2 threads per tile row, 32B (16 halves) each
  const int ldRow = t >> 1, ldHf = (t & 1) * 16;
  const __bf16* gA = A  + (size_t)(mBase + ldRow) * DIM + ldHf;
  const __bf16* gB = Bt + (size_t)(nBase + ldRow) * DIM + ldHf;
  const int wOffH = ldRow * LDH + ldHf;     // LDS write offset (halves)

  v8f acc[2][4];
#pragma unroll
  for (int mt = 0; mt < 2; ++mt)
#pragma unroll
    for (int nt = 0; nt < 4; ++nt)
      acc[mt][nt] = (v8f){0.f,0.f,0.f,0.f,0.f,0.f,0.f,0.f};

  // fragment LDS offsets (halves), per CDNA5 ISA 16-bit A/B layouts
  const int arow  = wm * 32 + (lane & 15);
  const int aoffh = (lane < 16) ? 0 : 8;    // A: kbase 0 / 8
  const int nrow  = wn * 64 + (lane & 15);
  const int boffh = (lane < 16) ? 0 : 8;    // B: kbase 0 / 8

#ifdef HAVE_ASYNC_LDS
  // ---- direct async global->LDS pipeline (no VGPR staging) ----
  async_cp16(gA,     &sA[0][0] + wOffH);
  async_cp16(gA + 8, &sA[0][0] + wOffH + 8);
  async_cp16(gB,     &sB[0][0] + wOffH);
  async_cp16(gB + 8, &sB[0][0] + wOffH + 8);

  for (int ks = 0; ks < 32; ++ks) {        // K = 1024, BK = 32
    WAIT_ASYNC0();                         // own async loads for buf[cur] complete
    __syncthreads();                       // everyone's loads landed; prior reads of buf[nxt] done

    if (ks < 31) {                         // overlap next K-slab with compute
      const __bf16* pA = gA + (ks + 1) * 32;
      const __bf16* pB = gB + (ks + 1) * 32;
      __bf16* dA = &sA[(ks + 1) & 1][0] + wOffH;
      __bf16* dB = &sB[(ks + 1) & 1][0] + wOffH;
      async_cp16(pA,     dA);
      async_cp16(pA + 8, dA + 8);
      async_cp16(pB,     dB);
      async_cp16(pB + 8, dB + 8);
    }

    const __bf16* sa = &sA[ks & 1][0];
    const __bf16* sb = &sB[ks & 1][0];
    v16bf af[2], bfr[4];
#pragma unroll
    for (int mt = 0; mt < 2; ++mt) {
      const __bf16* p = sa + (arow + mt * 16) * LDH + aoffh;
      af[mt] = cat8(*(const v8bf*)p, *(const v8bf*)(p + 16));
    }
#pragma unroll
    for (int nt = 0; nt < 4; ++nt) {
      const __bf16* p = sb + (nrow + nt * 16) * LDH + boffh;
      bfr[nt] = cat8(*(const v8bf*)p, *(const v8bf*)(p + 16));
    }
#pragma unroll
    for (int mt = 0; mt < 2; ++mt)
#pragma unroll
      for (int nt = 0; nt < 4; ++nt)
        acc[mt][nt] = __builtin_amdgcn_wmma_f32_16x16x32_bf16(
            false, af[mt], false, bfr[nt], (short)0, acc[mt][nt], false, false);
  }
#else
  // ---- fallback: register-staged double buffering ----
  uint4 ra0 = *(const uint4*)(gA);
  uint4 ra1 = *(const uint4*)(gA + 8);
  uint4 rb0 = *(const uint4*)(gB);
  uint4 rb1 = *(const uint4*)(gB + 8);
  {
    __bf16* wA = &sA[0][0] + wOffH;
    __bf16* wB = &sB[0][0] + wOffH;
    *(uint4*)wA = ra0; *(uint4*)(wA + 8) = ra1;
    *(uint4*)wB = rb0; *(uint4*)(wB + 8) = rb1;
  }
  __syncthreads();

  for (int ks = 0; ks < 32; ++ks) {        // K = 1024, BK = 32
    if (ks < 31) {                         // prefetch next K-slab into registers
      const __bf16* pA = gA + (ks + 1) * 32;
      const __bf16* pB = gB + (ks + 1) * 32;
      ra0 = *(const uint4*)(pA);
      ra1 = *(const uint4*)(pA + 8);
      rb0 = *(const uint4*)(pB);
      rb1 = *(const uint4*)(pB + 8);
    }

    const __bf16* sa = &sA[ks & 1][0];
    const __bf16* sb = &sB[ks & 1][0];
    v16bf af[2], bfr[4];
#pragma unroll
    for (int mt = 0; mt < 2; ++mt) {
      const __bf16* p = sa + (arow + mt * 16) * LDH + aoffh;
      af[mt] = cat8(*(const v8bf*)p, *(const v8bf*)(p + 16));
    }
#pragma unroll
    for (int nt = 0; nt < 4; ++nt) {
      const __bf16* p = sb + (nrow + nt * 16) * LDH + boffh;
      bfr[nt] = cat8(*(const v8bf*)p, *(const v8bf*)(p + 16));
    }
#pragma unroll
    for (int mt = 0; mt < 2; ++mt)
#pragma unroll
      for (int nt = 0; nt < 4; ++nt)
        acc[mt][nt] = __builtin_amdgcn_wmma_f32_16x16x32_bf16(
            false, af[mt], false, bfr[nt], (short)0, acc[mt][nt], false, false);

    if (ks < 31) {
      __bf16* dA = &sA[(ks + 1) & 1][0] + wOffH;
      __bf16* dB = &sB[(ks + 1) & 1][0] + wOffH;
      *(uint4*)dA = ra0; *(uint4*)(dA + 8) = ra1;
      *(uint4*)dB = rb0; *(uint4*)(dB + 8) = rb1;
      __syncthreads();
    }
  }
#endif

  // epilogue: + add, ReLU, store  (C/D layout: VGPR r -> M = r + (lane<16?0:8), N = lane&15)
  const int rowAdd = (lane < 16) ? 0 : 8;
#pragma unroll
  for (int mt = 0; mt < 2; ++mt) {
    int row0 = mBase + wm * 32 + mt * 16;
#pragma unroll
    for (int nt = 0; nt < 4; ++nt) {
      int col = nBase + wn * 64 + nt * 16 + (lane & 15);
      float addv = (MODE == 0) ? add[(size_t)(row0 >> 9) * DIM + col] : add[col];
#pragma unroll
      for (int r = 0; r < 8; ++r) {
        int row = row0 + r + rowAdd;
        float v = fmaxf(acc[mt][nt][r] + addv, 0.0f);
        if (MODE == 0) outH[(size_t)row * DIM + col] = (__bf16)v;
        else           outF[(size_t)row * DIM + col] = v;
      }
    }
  }
}

// ---------------- launcher ----------------
extern "C" void kernel_launch(void* const* d_in, const int* in_sizes, int n_in,
                              void* d_out, int out_size, void* d_ws, size_t ws_size,
                              hipStream_t stream) {
  const float* obj     = (const float*)d_in[0];
  const float* ctx     = (const float*)d_in[1];
  const int*   num_obj = (const int*)d_in[2];
  /* d_in[3] num_rels: shapes are static in the reference */
  const int*   pairs   = (const int*)d_in[4];
  const float* W1      = (const float*)d_in[5];
  const float* b1      = (const float*)d_in[6];
  const float* W2      = (const float*)d_in[7];
  const float* b2      = (const float*)d_in[8];
  float* out = (float*)d_out;

  char* ws = (char*)d_ws;
  __bf16* V    = (__bf16*)(ws + 0);            // 16384*1024*2  = 33,554,432
  __bf16* W1T  = (__bf16*)(ws + 33554432);     // 1024*1024*2   =  2,097,152
  __bf16* W2T  = (__bf16*)(ws + 35651584);     // 1024*1024*2   =  2,097,152
  float*  CtxC = (float*) (ws + 37748736);     // 32*1024*4     =    131,072
  __bf16* H    = (__bf16*)(ws + 37879808);     // 16384*1024*2  = 33,554,432

  // output 1: pass-through copy of object_branch_output
  copy_obj_kernel<<<3200, 256, 0, stream>>>((const float4*)obj, (float4*)out);

  // weights -> bf16, K-contiguous transposed layout
  transpose_to_bf16<<<dim3(16, 16), 256, 0, stream>>>(W1, W1T, 1024, 1024); // first 1024 rows of W1
  transpose_to_bf16<<<dim3(16, 16), 256, 0, stream>>>(W2, W2T, 1024, 1024);

  // per-(batch, col) context contribution + b1 (folds the ctx half of GEMM1)
  ctx_contrib_kernel<<<dim3(4, 32), 256, 0, stream>>>(ctx, W1, b1, CtxC);

  // pair gather + mean -> bf16
  gather_pair_mean<<<16384, 256, 0, stream>>>(obj, num_obj, pairs, V);

  // GEMM1: H = relu(V @ W1[:1024] + ctxc)   [bf16 out]
  gemm_bf16_wmma<0><<<dim3(128, 8), 256, 0, stream>>>(V, W1T, CtxC, H, nullptr);
  // GEMM2: out = relu(H @ W2 + b2)          [f32 out]
  gemm_bf16_wmma<1><<<dim3(128, 8), 256, 0, stream>>>(H, W2T, b2, nullptr, out + OUT_OFS);
}
